// Classifier_43645457662287
// MI455X (gfx1250) — matrix operand order repaired and checked by reading, not compiled, not run on hardware
//
#include <hip/hip_runtime.h>
#include <hip/hip_fp16.h>
#include <math.h>

typedef __attribute__((ext_vector_type(16))) _Float16 v16h;
typedef __attribute__((ext_vector_type(8)))  float    v8f;

// ---------------------------------------------------------------------------
// WMMA helpers (CDNA5 wave32, V_WMMA_F32_16X16X32_F16)
// A: 16(M)x32(K) f16, lane = M (both half-groups), elements map K per ISA:
//   lanes 0-15 : e0..7 -> K+0..7,  e8..15 -> K+16..23
//   lanes16-31 : e0..7 -> K+8..15, e8..15 -> K+24..31
// B: 32(K)x16(N) f16, lane = N (both half-groups), elements map K same as A
// C/D: v8f, element j -> row j + (lane>=16 ? 8 : 0), col = lane&15
// ---------------------------------------------------------------------------

static __device__ inline v16h load_a_g(const float* __restrict__ A, int lda, int k0,
                                       int kact, const float* __restrict__ abias,
                                       int mrem) {
  int lane = threadIdx.x & 31;
  int m = lane & 15; if (m >= mrem) m = 0;
  int kb = k0 + ((lane >> 4) << 3);
  const float* rp = A + (size_t)m * lda;
  v16h a;
#pragma unroll
  for (int e = 0; e < 8; ++e) {
    int k  = kb + e;
    int k2 = kb + 16 + e;
    float v  = (k  < kact) ? (rp[k]  + (abias ? abias[k]  : 0.0f)) : 0.0f;
    float v2 = (k2 < kact) ? (rp[k2] + (abias ? abias[k2] : 0.0f)) : 0.0f;
    a[e]     = (_Float16)v;
    a[e + 8] = (_Float16)v2;
  }
  return a;
}

static __device__ inline v16h load_a_l(const _Float16* A, int lda, int k0) {
  int lane = threadIdx.x & 31;
  int m = lane & 15;
  int kb = k0 + ((lane >> 4) << 3);
  const _Float16* rp = A + m * lda;
  v16h a;
#pragma unroll
  for (int e = 0; e < 8; ++e) {
    a[e]     = rp[kb + e];
    a[e + 8] = rp[kb + 16 + e];
  }
  return a;
}

// B from pre-converted f16 weights; rows >= kact are zero (K padding)
static __device__ inline v16h load_b_h(const _Float16* __restrict__ W, int ldb, int k0,
                                       int n0, int kact) {
  int lane = threadIdx.x & 31;
  int n = n0 + (lane & 15);
  int kb = k0 + ((lane >> 4) << 3);
  v16h b;
#pragma unroll
  for (int e = 0; e < 8; ++e) {
    int k = kb + e, k2 = kb + 16 + e;
    b[e]     = (k  < kact) ? W[(size_t)k  * ldb + n] : (_Float16)0.0f;
    b[e + 8] = (k2 < kact) ? W[(size_t)k2 * ldb + n] : (_Float16)0.0f;
  }
  return b;
}

static __device__ inline v8f wmma_acc(v16h a, v16h b, v8f c) {
  return __builtin_amdgcn_wmma_f32_16x16x32_f16(false, a, false, b, (short)0, c,
                                                false, false);
}

static __device__ inline v8f zero8() {
  v8f c = {0.f, 0.f, 0.f, 0.f, 0.f, 0.f, 0.f, 0.f};
  return c;
}

// A = f32 global (converted on the fly)
static __device__ inline v8f wmma_gemm_g(const float* A, int lda, const float* abias,
                                         const _Float16* W, int ldb,
                                         int kact, int n0, int mrem) {
  v8f c = zero8();
  int kpad = (kact + 31) & ~31;
  for (int k0 = 0; k0 < kpad; k0 += 32)
    c = wmma_acc(load_a_g(A, lda, k0, kact, abias, mrem), load_b_h(W, ldb, k0, n0, kact), c);
  return c;
}

// A = f16 global, K multiple of 32
static __device__ inline v8f wmma_gemm_gh(const _Float16* A, int lda, int kpad,
                                          const _Float16* W, int ldb, int kact,
                                          int n0, int mrem) {
  int lane = threadIdx.x & 31;
  int m = lane & 15; if (m >= mrem) m = 0;
  int kb0 = (lane >> 4) << 3;
  const _Float16* rp = A + (size_t)m * lda;
  v8f c = zero8();
  for (int k0 = 0; k0 < kpad; k0 += 32) {
    v16h a;
#pragma unroll
    for (int e = 0; e < 8; ++e) {
      a[e]     = rp[k0 + kb0 + e];
      a[e + 8] = rp[k0 + kb0 + 16 + e];
    }
    c = wmma_acc(a, load_b_h(W, ldb, k0, n0, kact), c);
  }
  return c;
}

// A = f16 LDS
static __device__ inline v8f wmma_gemm_l(const _Float16* A, int lda, int kpad,
                                         const _Float16* W, int ldb,
                                         int kact, int n0) {
  v8f c = zero8();
  for (int k0 = 0; k0 < kpad; k0 += 32)
    c = wmma_acc(load_a_l(A, lda, k0), load_b_h(W, ldb, k0, n0, kact), c);
  return c;
}

// Register-resident A fragments (reused across several N tiles)
template <int KF>
static __device__ inline void load_a_frags(v16h (&af)[KF], const float* A, int lda,
                                           const float* abias, int mrem) {
#pragma unroll
  for (int i = 0; i < KF; ++i)
    af[i] = load_a_g(A, lda, i * 32, KF * 32, abias, mrem);
}

template <int KF>
static __device__ inline v8f wmma_gemm_pre(const v16h (&af)[KF], const _Float16* W,
                                           int ldb, int kact, int n0) {
  v8f c = zero8();
#pragma unroll
  for (int i = 0; i < KF; ++i)
    c = wmma_acc(af[i], load_b_h(W, ldb, i * 32, n0, kact), c);
  return c;
}

static __device__ inline void store_lds_h(_Float16* D, int ldd, v8f c, int n0,
                                          const float* bias, int nact, bool dorelu) {
  int lane = threadIdx.x & 31;
  int n = n0 + (lane & 15);
  if (n >= nact) return;
  int mo = (lane >> 4) << 3;
  float bv = bias ? bias[n] : 0.0f;
#pragma unroll
  for (int j = 0; j < 8; ++j) {
    float v = c[j] + bv;
    if (dorelu) v = fmaxf(v, 0.0f);
    D[(mo + j) * ldd + n] = (_Float16)v;
  }
}

static __device__ inline void store_g_f32(float* D, int ldd, v8f c, int n0,
                                          const float* bias, int nact, bool dorelu,
                                          int mrem) {
  int lane = threadIdx.x & 31;
  int n = n0 + (lane & 15);
  if (n >= nact) return;
  int mo = (lane >> 4) << 3;
  float bv = bias ? bias[n] : 0.0f;
#pragma unroll
  for (int j = 0; j < 8; ++j) {
    int m = mo + j;
    if (m >= mrem) continue;
    float v = c[j] + bv;
    if (dorelu) v = fmaxf(v, 0.0f);
    D[(size_t)m * ldd + n] = v;
  }
}

static __device__ inline void store_g_h(_Float16* D, int ldd, v8f c, int n0,
                                        const float* bias, int nact, int mrem) {
  int lane = threadIdx.x & 31;
  int n = n0 + (lane & 15);
  if (n >= nact) return;
  int mo = (lane >> 4) << 3;
  float bv = bias ? bias[n] : 0.0f;
#pragma unroll
  for (int j = 0; j < 8; ++j) {
    int m = mo + j;
    if (m >= mrem) continue;
    D[(size_t)m * ldd + n] = (_Float16)(c[j] + bv);
  }
}

// ---------------------------------------------------------------------------
// Weight pre-conversion (once per launch; removes cvt from GEMM hot loops)
// ---------------------------------------------------------------------------

__global__ void k_cvt(const float* __restrict__ src, _Float16* dst, size_t n) {
  size_t i = (size_t)blockIdx.x * blockDim.x + threadIdx.x;
  if (i < n) dst[i] = (_Float16)src[i];
}

// copy rows x scols into rows x dcols, zero-padding cols >= scols
__global__ void k_cvt_pad(const float* __restrict__ src, _Float16* dst,
                          int rows, int scols, int dcols) {
  int i = blockIdx.x * blockDim.x + threadIdx.x;
  int r = i / dcols, c = i - r * dcols;
  if (r >= rows) return;
  dst[(size_t)r * dcols + c] =
      (c < scols) ? (_Float16)src[(size_t)r * scols + c] : (_Float16)0.0f;
}

// ---------------------------------------------------------------------------
// Small utility / graph kernels
// ---------------------------------------------------------------------------

__global__ void k_fill(float* p, size_t n) {
  size_t i = (size_t)blockIdx.x * blockDim.x + threadIdx.x;
  if (i < n) p[i] = 0.0f;
}

// logsumexp over 128 logits, one wave per node
__global__ void k_energy(const float* __restrict__ logits, float* __restrict__ e, int n) {
  int wv = (blockIdx.x * blockDim.x + threadIdx.x) >> 5;
  int lane = threadIdx.x & 31;
  if (wv >= n) return;
  const float* p = logits + (size_t)wv * 128;
  float v[4], m = -1e30f;
#pragma unroll
  for (int i = 0; i < 4; ++i) { v[i] = p[lane + 32 * i]; m = fmaxf(m, v[i]); }
  for (int off = 16; off; off >>= 1) m = fmaxf(m, __shfl_xor(m, off, 32));
  float s = 0.f;
#pragma unroll
  for (int i = 0; i < 4; ++i) s += __expf(v[i] - m);
  for (int off = 16; off; off >>= 1) s += __shfl_xor(s, off, 32);
  if (lane == 0) e[wv] = m + __logf(s);
}

__global__ void k_degree(const int* __restrict__ col, float* deg, int ne) {
  int i = blockIdx.x * blockDim.x + threadIdx.x;
  if (i >= ne) return;
  atomicAdd(&deg[col[i]], 1.0f);
}

__global__ void k_prop(const int* __restrict__ row, const int* __restrict__ col,
                       const float* __restrict__ deg, const float* __restrict__ e,
                       float* agg, int ne) {
  int i = blockIdx.x * blockDim.x + threadIdx.x;
  if (i >= ne) return;
  int r = row[i], c = col[i];
  atomicAdd(&agg[c], e[r] / deg[c]);   // deg[c] >= 1 for any existing edge
}

__global__ void k_combine(float* e, const float* agg, int n) {
  int i = blockIdx.x * blockDim.x + threadIdx.x;
  if (i >= n) return;
  e[i] = 0.5f * e[i] + 0.5f * agg[i];  // ALPHA = 0.5
}

__global__ void k_stats(const float* __restrict__ e, float* stats, int n) {
  int i = blockIdx.x * blockDim.x + threadIdx.x;
  float v = (i < n) ? e[i] : 0.0f;
  float v2 = v * v;
  for (int off = 16; off; off >>= 1) {
    v  += __shfl_xor(v,  off, 32);
    v2 += __shfl_xor(v2, off, 32);
  }
  if ((threadIdx.x & 31) == 0) {
    atomicAdd(&stats[0], v);
    atomicAdd(&stats[1], v2);
  }
}

__global__ void k_enorm(float* e, const float* __restrict__ stats, int n) {
  int i = blockIdx.x * blockDim.x + threadIdx.x;
  if (i >= n) return;
  float s1 = stats[0], s2 = stats[1];
  float mean = s1 / (float)n;
  float var = (s2 - s1 * s1 / (float)n) / (float)(n - 1);  // ddof=1
  float sd = sqrtf(fmaxf(var, 0.0f));
  e[i] = (e[i] - mean) / (sd + 1e-6f);
}

// energy MLP 1->16->8, one thread per node (tiny)
__global__ void k_pe(const float* __restrict__ e, const float* __restrict__ w1,
                     const float* __restrict__ b1, const float* __restrict__ w2,
                     const float* __restrict__ b2, float* pe, int n) {
  int i = blockIdx.x * blockDim.x + threadIdx.x;
  if (i >= n) return;
  float v = e[i];
  float acc[8];
#pragma unroll
  for (int j = 0; j < 8; ++j) acc[j] = b2[j];
#pragma unroll
  for (int k = 0; k < 16; ++k) {
    float t = fmaxf(v * w1[k] + b1[k], 0.0f);
#pragma unroll
    for (int j = 0; j < 8; ++j) acc[j] += t * w2[k * 8 + j];
  }
#pragma unroll
  for (int j = 0; j < 8; ++j) pe[(size_t)i * 8 + j] = acc[j];
}

// GCN self-loop term: g[n][k] = h[n][k] / (deg[n]+1)   (dis[n]^2 = 1/deg2[n])
__global__ void k_gself(const float* __restrict__ h, const float* __restrict__ deg,
                        float* g, int n) {
  size_t idx = (size_t)blockIdx.x * blockDim.x + threadIdx.x;
  if (idx >= (size_t)n * 128) return;
  int node = (int)(idx >> 7);
  g[idx] = h[idx] / (deg[node] + 1.0f);
}

// GCN edge scatter: g[c] += dis[r]*dis[c]*h[r]; 32 threads/edge, 4 floats/thread
__global__ void k_gscatter(const int* __restrict__ row, const int* __restrict__ col,
                           const float* __restrict__ deg, const float* __restrict__ h,
                           float* g, int ne) {
  long long gid = (long long)blockIdx.x * blockDim.x + threadIdx.x;
  long long i = gid >> 5;
  if (i >= ne) return;
  int kq = ((int)gid & 31) << 2;
  int r = row[i], c = col[i];
  float w = rsqrtf((deg[r] + 1.0f) * (deg[c] + 1.0f));
  const float* hp = h + (size_t)r * 128 + kq;
  float* gp = g + (size_t)c * 128 + kq;
  atomicAdd(gp + 0, w * hp[0]);
  atomicAdd(gp + 1, w * hp[1]);
  atomicAdd(gp + 2, w * hp[2]);
  atomicAdd(gp + 3, w * hp[3]);
}

// ---------------------------------------------------------------------------
// Fused dense chains (16-node tiles, 8 waves, WMMA everywhere)
// ---------------------------------------------------------------------------

// x(256) -> relu(512) -> px(256) -> h = px @ gcn_w (128); only h hits HBM
__global__ __launch_bounds__(256)
void k_xchain(const float* __restrict__ x,
              const _Float16* __restrict__ fp_w1h, const float* __restrict__ fp_b1,
              const _Float16* __restrict__ fp_w2h, const float* __restrict__ fp_b2,
              const _Float16* __restrict__ gcn_wh, float* h, int n) {
  __shared__ _Float16 t1l[16 * 512];
  __shared__ _Float16 pxl[16 * 256];
  int node0 = blockIdx.x * 16;
  int mrem = n - node0; if (mrem > 16) mrem = 16;
  int wv = threadIdx.x >> 5;
  // A tile (16x256 of x) resident in registers, reused across 4 N-tiles/wave
  v16h axf[8];
  load_a_frags<8>(axf, x + (size_t)node0 * 256, 256, nullptr, mrem);
  for (int t = wv; t < 32; t += 8) {
    v8f c = wmma_gemm_pre<8>(axf, fp_w1h, 512, 256, t * 16);
    store_lds_h(t1l, 512, c, t * 16, fp_b1, 512, true);
  }
  __syncthreads();
  for (int t = wv; t < 16; t += 8) {
    v8f c = wmma_gemm_l(t1l, 512, 512, fp_w2h, 256, 512, t * 16);
    store_lds_h(pxl, 256, c, t * 16, fp_b2, 256, false);
  }
  __syncthreads();
  for (int t = wv; t < 8; t += 8) {
    v8f c = wmma_gemm_l(pxl, 256, 256, gcn_wh, 128, 256, t * 16);
    store_g_f32(h + (size_t)node0 * 128, 128, c, t * 16, nullptr, 128, false, mrem);
  }
}

// logits(128) -> relu(128) -> pl(64), pl stored f16
__global__ __launch_bounds__(256)
void k_lchain(const float* __restrict__ logits,
              const _Float16* __restrict__ lp_w1h, const float* __restrict__ lp_b1,
              const _Float16* __restrict__ lp_w2h, const float* __restrict__ lp_b2,
              _Float16* pl, int n) {
  __shared__ _Float16 tl[16 * 128];
  int node0 = blockIdx.x * 16;
  int mrem = n - node0; if (mrem > 16) mrem = 16;
  int wv = threadIdx.x >> 5;
  const float* A = logits + (size_t)node0 * 128;
  for (int t = wv; t < 8; t += 8) {
    v8f c = wmma_gemm_g(A, 128, nullptr, lp_w1h, 128, 128, t * 16, mrem);
    store_lds_h(tl, 128, c, t * 16, lp_b1, 128, true);
  }
  __syncthreads();
  for (int t = wv; t < 4; t += 8) {
    v8f c = wmma_gemm_l(tl, 128, 128, lp_w2h, 64, 128, t * 16);
    store_g_h(pl + (size_t)node0 * 64, 64, c, t * 16, lp_b2, 64, mrem);
  }
}

// g(+gcn_b) -> gx(256) -> xt/lt/et(193) -> attention fuse -> h1(256) -> h2(128) -> pred
__global__ __launch_bounds__(256)
void k_tail(const float* __restrict__ g, const float* __restrict__ gcn_b,
            const _Float16* __restrict__ gt_wh, const float* __restrict__ gt_b,
            const _Float16* __restrict__ pl, const float* __restrict__ pe,
            const _Float16* __restrict__ af_xwh, const float* __restrict__ af_xb,
            const _Float16* __restrict__ af_lwh, const float* __restrict__ af_lb,
            const _Float16* __restrict__ af_ewh, const float* __restrict__ af_eb,
            const float* __restrict__ af_aw, const float* __restrict__ af_ab,
            const _Float16* __restrict__ fu_w1h, const float* __restrict__ fu_b1,
            const _Float16* __restrict__ fu_w2h, const float* __restrict__ fu_b2,
            const float* __restrict__ fu_w3, const float* __restrict__ fu_b3,
            float* pred, int n) {
  __shared__ _Float16 gxl[16 * 256];
  __shared__ _Float16 xtl[16 * 208];
  __shared__ _Float16 ltl[16 * 208];
  __shared__ _Float16 etl[16 * 208];
  __shared__ _Float16 fsl[16 * 224];   // fused, zero-padded K 193..223
  __shared__ _Float16 h1l[16 * 256];
  __shared__ _Float16 h2l[16 * 128];
  int node0 = blockIdx.x * 16;
  int mrem = n - node0; if (mrem > 16) mrem = 16;
  int wv = threadIdx.x >> 5, lane = threadIdx.x & 31;

  for (int idx = threadIdx.x; idx < 16 * 224; idx += 256) fsl[idx] = (_Float16)0.0f;

  // gx = (g + gcn_b) @ gt_w + gt_b   (K=128 -> 256); A frags register-resident
  {
    v16h agf[4];
    load_a_frags<4>(agf, g + (size_t)node0 * 128, 128, gcn_b, mrem);
    for (int t = wv; t < 16; t += 8) {
      v8f c = wmma_gemm_pre<4>(agf, gt_wh, 256, 128, t * 16);
      store_lds_h(gxl, 256, c, t * 16, gt_b, 256, false);
    }
  }
  __syncthreads();
  // xt = gx @ af_xw + af_xb   (256 -> 193, weights padded to 208 cols)
  for (int t = wv; t < 13; t += 8) {
    v8f c = wmma_gemm_l(gxl, 256, 256, af_xwh, 208, 256, t * 16);
    store_lds_h(xtl, 208, c, t * 16, af_xb, 193, false);
  }
  // lt = pl @ af_lw + af_lb   (64 -> 193)
  for (int t = wv; t < 13; t += 8) {
    v8f c = wmma_gemm_gh(pl + (size_t)node0 * 64, 64, 64, af_lwh, 208, 64,
                         t * 16, mrem);
    store_lds_h(ltl, 208, c, t * 16, af_lb, 193, false);
  }
  // et = pe @ af_ew + af_eb   (8 -> 193, K zero-padded to 32)
  for (int t = wv; t < 13; t += 8) {
    v8f c = wmma_gemm_g(pe + (size_t)node0 * 8, 8, nullptr, af_ewh, 208, 8,
                        t * 16, mrem);
    store_lds_h(etl, 208, c, t * 16, af_eb, 193, false);
  }
  __syncthreads();

  // attention weights + fused average, one wave per row
  for (int m = wv; m < 16; m += 8) {
    float dx = 0.f, dl = 0.f, de = 0.f;
    for (int j = lane; j < 193; j += 32) {
      float w = af_aw[j];
      dx += (float)xtl[m * 208 + j] * w;
      dl += (float)ltl[m * 208 + j] * w;
      de += (float)etl[m * 208 + j] * w;
    }
    for (int off = 16; off; off >>= 1) {
      dx += __shfl_xor(dx, off, 32);
      dl += __shfl_xor(dl, off, 32);
      de += __shfl_xor(de, off, 32);
    }
    float ab = af_ab[0];
    float xw = 1.f / (1.f + __expf(-(dx + ab)));
    float lw = 1.f / (1.f + __expf(-(dl + ab)));
    float ew = 1.f / (1.f + __expf(-(de + ab)));
    float inv = 1.f / (xw + lw + ew);
    for (int j = lane; j < 193; j += 32) {
      float f = (xw * (float)xtl[m * 208 + j] + lw * (float)ltl[m * 208 + j] +
                 ew * (float)etl[m * 208 + j]) * inv;
      fsl[m * 224 + j] = (_Float16)f;
    }
  }
  __syncthreads();

  // h1 = relu(fused @ fu_w1 + fu_b1)   (193 -> 256)
  for (int t = wv; t < 16; t += 8) {
    v8f c = wmma_gemm_l(fsl, 224, 224, fu_w1h, 256, 193, t * 16);
    store_lds_h(h1l, 256, c, t * 16, fu_b1, 256, true);
  }
  __syncthreads();
  // h2 = relu(h1 @ fu_w2 + fu_b2)   (256 -> 128)
  for (int t = wv; t < 8; t += 8) {
    v8f c = wmma_gemm_l(h1l, 256, 256, fu_w2h, 128, 256, t * 16);
    store_lds_h(h2l, 128, c, t * 16, fu_b2, 128, true);
  }
  __syncthreads();
  // pred = sigmoid(h2 . fu_w3 + fu_b3)
  for (int m = wv; m < 16; m += 8) {
    float d = 0.f;
    for (int j = lane; j < 128; j += 32) d += (float)h2l[m * 128 + j] * fu_w3[j];
    for (int off = 16; off; off >>= 1) d += __shfl_xor(d, off, 32);
    if (lane == 0 && m < mrem)
      pred[node0 + m] = 1.f / (1.f + __expf(-(d + fu_b3[0])));
  }
}

__global__ void k_gather(const float* __restrict__ pred, const int* __restrict__ mask,
                         float* out, int m) {
  int i = blockIdx.x * blockDim.x + threadIdx.x;
  if (i >= m) return;
  out[i] = pred[mask[i]];
}

// ---------------------------------------------------------------------------
// Host launcher
// ---------------------------------------------------------------------------

extern "C" void kernel_launch(void* const* d_in, const int* in_sizes, int n_in,
                              void* d_out, int out_size, void* d_ws, size_t ws_size,
                              hipStream_t stream) {
  (void)n_in; (void)ws_size;
  const float* logits = (const float*)d_in[0];
  const float* x      = (const float*)d_in[1];
  const int*   eidx   = (const int*)d_in[2];
  const int*   mask   = (const int*)d_in[3];
  const float* fp_w1 = (const float*)d_in[4];
  const float* fp_b1 = (const float*)d_in[5];
  const float* fp_w2 = (const float*)d_in[6];
  const float* fp_b2 = (const float*)d_in[7];
  const float* ep_w1 = (const float*)d_in[8];
  const float* ep_b1 = (const float*)d_in[9];
  const float* ep_w2 = (const float*)d_in[10];
  const float* ep_b2 = (const float*)d_in[11];
  const float* lp_w1 = (const float*)d_in[12];
  const float* lp_b1 = (const float*)d_in[13];
  const float* lp_w2 = (const float*)d_in[14];
  const float* lp_b2 = (const float*)d_in[15];
  const float* gcn_w = (const float*)d_in[16];
  const float* gcn_b = (const float*)d_in[17];
  const float* gt_w  = (const float*)d_in[18];
  const float* gt_b  = (const float*)d_in[19];
  const float* af_xw = (const float*)d_in[20];
  const float* af_xb = (const float*)d_in[21];
  const float* af_lw = (const float*)d_in[22];
  const float* af_lb = (const float*)d_in[23];
  const float* af_ew = (const float*)d_in[24];
  const float* af_eb = (const float*)d_in[25];
  const float* af_aw = (const float*)d_in[26];
  const float* af_ab = (const float*)d_in[27];
  const float* fu_w1 = (const float*)d_in[28];
  const float* fu_b1 = (const float*)d_in[29];
  const float* fu_w2 = (const float*)d_in[30];
  const float* fu_b2 = (const float*)d_in[31];
  const float* fu_w3 = (const float*)d_in[32];
  const float* fu_b3 = (const float*)d_in[33];

  const int C = 128;
  const int N = in_sizes[0] / C;
  const int E = in_sizes[2] / 2;
  const int* row = eidx;
  const int* col = eidx + E;

  char* ws = (char*)d_ws;
  size_t off = 0;
  auto carve = [&](size_t bytes) -> void* {
    void* p = (void*)(ws + off);
    off += (bytes + 255) & ~(size_t)255;
    return p;
  };
  float*     e     = (float*)carve((size_t)N * 4);
  float*     agg   = (float*)carve((size_t)N * 4);
  float*     deg   = (float*)carve((size_t)N * 4);
  float*     stats = (float*)carve(256);
  float*     pe    = (float*)carve((size_t)N * 8 * 4);
  _Float16*  pl    = (_Float16*)carve((size_t)N * 64 * 2);
  float*     h     = (float*)carve((size_t)N * 128 * 4);
  float*     g     = (float*)carve((size_t)N * 128 * 4);
  float*     pred  = (float*)carve((size_t)N * 4);
  // f16 weight copies
  _Float16* fp_w1h = (_Float16*)carve((size_t)in_sizes[4] * 2);
  _Float16* fp_w2h = (_Float16*)carve((size_t)in_sizes[6] * 2);
  _Float16* lp_w1h = (_Float16*)carve((size_t)in_sizes[12] * 2);
  _Float16* lp_w2h = (_Float16*)carve((size_t)in_sizes[14] * 2);
  _Float16* gcn_wh = (_Float16*)carve((size_t)in_sizes[16] * 2);
  _Float16* gt_wh  = (_Float16*)carve((size_t)in_sizes[18] * 2);
  _Float16* fu_w1h = (_Float16*)carve((size_t)in_sizes[28] * 2);
  _Float16* fu_w2h = (_Float16*)carve((size_t)in_sizes[30] * 2);
  _Float16* af_xwh = (_Float16*)carve((size_t)256 * 208 * 2);  // 256x193 -> 256x208
  _Float16* af_lwh = (_Float16*)carve((size_t)64 * 208 * 2);   // 64x193  -> 64x208
  _Float16* af_ewh = (_Float16*)carve((size_t)8 * 208 * 2);    // 8x193   -> 8x208

  auto cdiv = [](long long a, long long b) { return (int)((a + b - 1) / b); };

  // weight conversion (cheap, once per launch)
  k_cvt<<<cdiv(in_sizes[4], 256), 256, 0, stream>>>(fp_w1, fp_w1h, (size_t)in_sizes[4]);
  k_cvt<<<cdiv(in_sizes[6], 256), 256, 0, stream>>>(fp_w2, fp_w2h, (size_t)in_sizes[6]);
  k_cvt<<<cdiv(in_sizes[12], 256), 256, 0, stream>>>(lp_w1, lp_w1h, (size_t)in_sizes[12]);
  k_cvt<<<cdiv(in_sizes[14], 256), 256, 0, stream>>>(lp_w2, lp_w2h, (size_t)in_sizes[14]);
  k_cvt<<<cdiv(in_sizes[16], 256), 256, 0, stream>>>(gcn_w, gcn_wh, (size_t)in_sizes[16]);
  k_cvt<<<cdiv(in_sizes[18], 256), 256, 0, stream>>>(gt_w, gt_wh, (size_t)in_sizes[18]);
  k_cvt<<<cdiv(in_sizes[28], 256), 256, 0, stream>>>(fu_w1, fu_w1h, (size_t)in_sizes[28]);
  k_cvt<<<cdiv(in_sizes[30], 256), 256, 0, stream>>>(fu_w2, fu_w2h, (size_t)in_sizes[30]);
  k_cvt_pad<<<cdiv(256 * 208, 256), 256, 0, stream>>>(af_xw, af_xwh, 256, 193, 208);
  k_cvt_pad<<<cdiv(64 * 208, 256), 256, 0, stream>>>(af_lw, af_lwh, 64, 193, 208);
  k_cvt_pad<<<cdiv(8 * 208, 256), 256, 0, stream>>>(af_ew, af_ewh, 8, 193, 208);

  k_fill<<<cdiv(N, 256), 256, 0, stream>>>(deg, (size_t)N);
  k_fill<<<1, 256, 0, stream>>>(stats, (size_t)2);
  k_energy<<<cdiv(N, 8), 256, 0, stream>>>(logits, e, N);
  k_degree<<<cdiv(E, 256), 256, 0, stream>>>(col, deg, E);
  for (int it = 0; it < 2; ++it) {   // PROP_LAYERS = 2
    k_fill<<<cdiv(N, 256), 256, 0, stream>>>(agg, (size_t)N);
    k_prop<<<cdiv(E, 256), 256, 0, stream>>>(row, col, deg, e, agg, E);
    k_combine<<<cdiv(N, 256), 256, 0, stream>>>(e, agg, N);
  }
  k_stats<<<cdiv(N, 256), 256, 0, stream>>>(e, stats, N);
  k_enorm<<<cdiv(N, 256), 256, 0, stream>>>(e, stats, N);
  k_pe<<<cdiv(N, 256), 256, 0, stream>>>(e, ep_w1, ep_b1, ep_w2, ep_b2, pe, N);

  int nt = cdiv(N, 16);
  k_xchain<<<nt, 256, 0, stream>>>(x, fp_w1h, fp_b1, fp_w2h, fp_b2, gcn_wh, h, N);
  k_lchain<<<nt, 256, 0, stream>>>(logits, lp_w1h, lp_b1, lp_w2h, lp_b2, pl, N);
  k_gself<<<cdiv((long long)N * 128, 256), 256, 0, stream>>>(h, deg, g, N);
  k_gscatter<<<cdiv((long long)E * 32, 256), 256, 0, stream>>>(row, col, deg, h, g, E);
  k_tail<<<nt, 256, 0, stream>>>(g, gcn_b, gt_wh, gt_b, pl, pe,
                                 af_xwh, af_xb, af_lwh, af_lb, af_ewh, af_eb,
                                 af_aw, af_ab, fu_w1h, fu_b1, fu_w2h, fu_b2,
                                 fu_w3, fu_b3, pred, N);
  k_gather<<<cdiv(out_size, 256), 256, 0, stream>>>(pred, mask, (float*)d_out, out_size);
}
